// TiSASMultiHeadAttention_35141422416209
// MI455X (gfx1250) — matrix operand order repaired and checked by reading, not compiled, not run on hardware
//
#include <hip/hip_runtime.h>
#include <hip/hip_bf16.h>

typedef __attribute__((ext_vector_type(16))) _Float16 v16h;
typedef __attribute__((ext_vector_type(8)))  float    v8f;
typedef __attribute__((ext_vector_type(4)))  float    f32x4;

#define HIDDEN 128
#define LSEQ   512
#define BATCH  2
#define BL     (BATCH * LSEQ)      // 1024
#define SCPAD  520                 // 512 + 8 pad: kills 8-way LDS bank conflict

union H2U { _Float16 h[2]; unsigned int u; };
union AV  { v16h v; unsigned int u[8]; };

// ---------------------------------------------------------------------------
// dst[BLx128] = X[BLx128] @ W[128x128] + bias[128] (+ extra[BLx128] if given)
// grid = BL/16 = 64 workgroups, 256 threads (8 waves). Wave w owns the 16-col
// tile n0 = 16*w; each wave does 4x v_wmma_f32_16x16x32_f16 over K=128.
// ---------------------------------------------------------------------------
__global__ __launch_bounds__(256)
void gemm16_wmma(const float* __restrict__ X, const float* __restrict__ W,
                 const float* __restrict__ bias, const float* __restrict__ extra,
                 float* __restrict__ dst)
{
    __shared__ _Float16 xl[16 * HIDDEN];     // 16-row A tile as f16 (4 KB)
    const int tid  = threadIdx.x;
    const int wave = tid >> 5;
    const int lane = tid & 31;
    const int m0   = blockIdx.x * 16;

    // cooperative load of the 16x128 X tile, f32 -> f16 (coalesced)
#pragma unroll
    for (int e = 0; e < 8; ++e) {
        int i = tid + 256 * e;               // 0..2047
        xl[i] = (_Float16)X[m0 * HIDDEN + i];
    }
    __syncthreads();

    const int n0    = wave * 16;
    const int l15   = lane & 15;
    const int hi    = lane >> 4;             // 0 = low lane half, 1 = high
    const int row_a = l15;                   // A: M = lane&15 for both halves
    const int koffA = hi * 8;                // A: K halves 0-7/16-23 vs 8-15/24-31
    const int nb    = n0 + l15;              // B/C/D column
    const int kselB = hi * 16;               // unused; B uses A's K order

    // C init = bias (depends on column only -> same for all 8 C rows)
    v8f c;
    float bcol = bias[nb];
#pragma unroll
    for (int j = 0; j < 8; ++j) c[j] = bcol;

    const unsigned int* xlu = (const unsigned int*)xl;

#pragma unroll
    for (int kb = 0; kb < HIDDEN; kb += 32) {
        AV a, b;
#pragma unroll
        for (int j = 0; j < 8; ++j) {
            // A fragment (16-bit A 16x32 layout, ISA 7.12.2): VGPR j holds
            // packed K = base, base+1
            int kk = kb + ((j < 4) ? 0 : 16) + koffA + 2 * (j & 3);
            a.u[j] = xlu[(row_a * HIDDEN + kk) >> 1];
            // B fragment: same K order as A (halves 0-7/16-23 vs 8-15/24-31),
            // for column nb.  Lanes 0..15 read 16 consecutive
            // columns of one W row -> coalesced 64B.
            int kw = kb + ((j < 4) ? 0 : 16) + koffA + 2 * (j & 3);
            H2U p;
            p.h[0] = (_Float16)W[(kw + 0) * HIDDEN + nb];
            p.h[1] = (_Float16)W[(kw + 1) * HIDDEN + nb];
            b.u[j] = p.u;
        }
        c = __builtin_amdgcn_wmma_f32_16x16x32_f16(
                /*neg_a=*/false, a.v, /*neg_b=*/false, b.v,
                /*c_mod=*/(short)0, c, /*reuse_a=*/false, /*reuse_b=*/false);
    }

    // D layout: VGPR j -> row m0 + j (+8 for high lane half), col nb
#pragma unroll
    for (int j = 0; j < 8; ++j) {
        int row = m0 + j + hi * 8;
        float v = c[j];
        if (extra) v += extra[row * HIDDEN + nb];
        dst[row * HIDDEN + nb] = v;
    }
}

// ---------------------------------------------------------------------------
// Core attention: one workgroup per (b, qpos). 8 waves x 32 lanes.
// Lane l owns hidden dims 4l..4l+3  (head h = l/4, dims 4(l%4)..).
// Phase 1: stream time_k[b,q] (256 KB, NT) fused with k' -> scores.
// Phase 2: per-head softmax (one wave per head).
// Phase 3: stream time_v[b,q] fused with v' -> ctx, deterministic reduce.
// ---------------------------------------------------------------------------
__global__ __launch_bounds__(256)
void tisas_attn(const float* __restrict__ qbuf, const float* __restrict__ kpbuf,
                const float* __restrict__ vpbuf, const float* __restrict__ time_k,
                const float* __restrict__ time_v, const float* __restrict__ amask,
                float* __restrict__ ctxbuf)
{
    __shared__ float sc[8 * SCPAD];          // scores/probs per head (16.6 KB)
    __shared__ float part[8][HIDDEN];        // per-wave ctx partials (4 KB)

    const int tid  = threadIdx.x;
    const int wave = tid >> 5;
    const int lane = tid & 31;
    const int bq   = blockIdx.x;             // b*L + qpos
    const int b    = bq >> 9;
    const int qpos = bq & 511;
    const int h    = lane >> 2;              // head owned by this lane group

    // query vector: each lane keeps its 4 dims
    f32x4 qv = *(const f32x4*)(qbuf + (size_t)bq * HIDDEN + 4 * lane);

    const float* tkbase = time_k + (size_t)bq * LSEQ * HIDDEN;
    const float* tvbase = time_v + (size_t)bq * LSEQ * HIDDEN;

    // -------- Phase 1: scores[h][k] = q . (k'(k) + time_k(q,k)) --------
    for (int k = wave; k < LSEQ; k += 8) {
        f32x4 t  = __builtin_nontemporal_load(
                       (const f32x4*)(tkbase + (size_t)k * HIDDEN + 4 * lane));
        f32x4 kp = *(const f32x4*)(kpbuf + ((size_t)(b * LSEQ + k)) * HIDDEN + 4 * lane);
        float p = qv.x * (t.x + kp.x) + qv.y * (t.y + kp.y)
                + qv.z * (t.z + kp.z) + qv.w * (t.w + kp.w);
        p += __shfl_xor(p, 1, 32);           // reduce 4-lane group (one head)
        p += __shfl_xor(p, 2, 32);
        if ((lane & 3) == 0) sc[h * SCPAD + k] = p;
    }
    __syncthreads();

    // -------- Phase 2: softmax; wave w handles head w --------
    {
        const float* mrow = amask + (size_t)b * LSEQ * LSEQ + (size_t)qpos * LSEQ;
        float vals[16];
        float mx = -3.4e38f;
#pragma unroll
        for (int i = 0; i < 16; ++i) {
            int k = lane + 32 * i;
            float t = sc[wave * SCPAD + k] * 0.25f + mrow[k];  // /sqrt(16) + mask
            vals[i] = t;
            mx = fmaxf(mx, t);
        }
        for (int o = 1; o < 32; o <<= 1) mx = fmaxf(mx, __shfl_xor(mx, o, 32));
        float sum = 0.f;
#pragma unroll
        for (int i = 0; i < 16; ++i) {
            float e = __expf(vals[i] - mx);
            vals[i] = e;
            sum += e;
        }
        for (int o = 1; o < 32; o <<= 1) sum += __shfl_xor(sum, o, 32);
        float inv = 1.0f / sum;
#pragma unroll
        for (int i = 0; i < 16; ++i)
            sc[wave * SCPAD + lane + 32 * i] = vals[i] * inv;
    }
    __syncthreads();

    // -------- Phase 3: ctx[hd] = sum_k probs[h,k] * (v'(k) + time_v(q,k)) ----
    f32x4 acc = {0.f, 0.f, 0.f, 0.f};
    for (int k = wave; k < LSEQ; k += 8) {
        f32x4 t  = __builtin_nontemporal_load(
                       (const f32x4*)(tvbase + (size_t)k * HIDDEN + 4 * lane));
        f32x4 vp = *(const f32x4*)(vpbuf + ((size_t)(b * LSEQ + k)) * HIDDEN + 4 * lane);
        float p = sc[h * SCPAD + k];         // padded stride -> no bank conflict
        acc.x += p * (t.x + vp.x);
        acc.y += p * (t.y + vp.y);
        acc.z += p * (t.z + vp.z);
        acc.w += p * (t.w + vp.w);
    }
    part[wave][4 * lane + 0] = acc.x;
    part[wave][4 * lane + 1] = acc.y;
    part[wave][4 * lane + 2] = acc.z;
    part[wave][4 * lane + 3] = acc.w;
    __syncthreads();

    if (tid < HIDDEN) {                      // fixed-order (deterministic) reduce
        float s = 0.f;
#pragma unroll
        for (int w = 0; w < 8; ++w) s += part[w][tid];
        ctxbuf[(size_t)bq * HIDDEN + tid] = s;
    }
}

// ---------------------------------------------------------------------------
extern "C" void kernel_launch(void* const* d_in, const int* in_sizes, int n_in,
                              void* d_out, int out_size, void* d_ws, size_t ws_size,
                              hipStream_t stream) {
    (void)in_sizes; (void)n_in; (void)out_size; (void)ws_size;

    const float* x       = (const float*)d_in[0];
    const float* time_k  = (const float*)d_in[1];
    const float* time_v  = (const float*)d_in[2];
    const float* abs_k   = (const float*)d_in[3];
    const float* abs_v   = (const float*)d_in[4];
    const float* amask   = (const float*)d_in[5];
    const float* Wq      = (const float*)d_in[6];
    const float* bq      = (const float*)d_in[7];
    const float* Wk      = (const float*)d_in[8];
    const float* bk      = (const float*)d_in[9];
    const float* Wv      = (const float*)d_in[10];
    const float* bv      = (const float*)d_in[11];
    const float* Wo      = (const float*)d_in[12];
    const float* bo      = (const float*)d_in[13];

    float* ws     = (float*)d_ws;
    float* qbuf   = ws;                       // 1024*128
    float* kpbuf  = ws + 1 * BL * HIDDEN;     // k' = xWk + bk + abs_k
    float* vpbuf  = ws + 2 * BL * HIDDEN;     // v' = xWv + bv + abs_v
    float* ctxbuf = ws + 3 * BL * HIDDEN;

    dim3 blk(256);

    gemm16_wmma<<<BL / 16, blk, 0, stream>>>(x, Wq, bq, nullptr, qbuf);
    gemm16_wmma<<<BL / 16, blk, 0, stream>>>(x, Wk, bk, abs_k, kpbuf);
    gemm16_wmma<<<BL / 16, blk, 0, stream>>>(x, Wv, bv, abs_v, vpbuf);

    tisas_attn<<<BL, blk, 0, stream>>>(qbuf, kpbuf, vpbuf, time_k, time_v,
                                       amask, ctxbuf);

    gemm16_wmma<<<BL / 16, blk, 0, stream>>>(ctxbuf, Wo, bo, nullptr,
                                             (float*)d_out);
}
